// SingleHeadQKV_45904610460159
// MI455X (gfx1250) — matrix operand order repaired and checked by reading, not compiled, not run on hardware
//
#include <hip/hip_runtime.h>
#include <hip/hip_bf16.h>

typedef __bf16 bf16_t;
typedef __attribute__((ext_vector_type(16))) __bf16 v16bf;
typedef __attribute__((ext_vector_type(8)))  float  v8f;

#define T_SEQ 4096
#define CIN   1024
#define NB    2

union Frag {            // one WMMA 16-bit operand = two 128-bit loads
  v16bf v;
  uint4 q[2];
};

__device__ __forceinline__ float silu_f(float x) {
  return x / (1.0f + __expf(-x));
}

// ---------------------------------------------------------------------------
// Kernel 0a: X f32 -> bf16, row-major (vectorized 4-wide)
// ---------------------------------------------------------------------------
__global__ __launch_bounds__(256) void xcvt_kernel(
    const float* __restrict__ X, bf16_t* __restrict__ Xb)
{
  size_t i = ((size_t)blockIdx.x * 256 + threadIdx.x) * 4;
  float4 x = *(const float4*)(X + i);
  union { bf16_t b[4]; uint2 u; } o;
  o.b[0] = (bf16_t)x.x; o.b[1] = (bf16_t)x.y;
  o.b[2] = (bf16_t)x.z; o.b[3] = (bf16_t)x.w;
  *(uint2*)(Xb + i) = o.u;
}

// ---------------------------------------------------------------------------
// Kernel 0b: transpose [rows,cols] -> [cols,rows] with cvt to bf16
// ---------------------------------------------------------------------------
template <typename SRC>
__global__ __launch_bounds__(256) void trans_kernel(
    const SRC* __restrict__ W, bf16_t* __restrict__ WT, int rows, int cols)
{
  __shared__ bf16_t tile[32][33];
  const int tI = blockIdx.x % (rows / 32);
  const int tJ = blockIdx.x / (rows / 32);
  const int c  = threadIdx.x & 31;
  const int r0 = threadIdx.x >> 5;
#pragma unroll
  for (int i = 0; i < 4; ++i) {
    int r = r0 + 8 * i;
    tile[r][c] = (bf16_t)W[(size_t)(tI * 32 + r) * cols + tJ * 32 + c];
  }
  __syncthreads();
#pragma unroll
  for (int i = 0; i < 4; ++i) {
    int r = r0 + 8 * i;
    WT[(size_t)(tJ * 32 + r) * rows + tI * 32 + c] = tile[c][r];
  }
}

// ---------------------------------------------------------------------------
// Kernel 1: fused QKV projection, software-pipelined bf16 GEMM.
//   Per-wave tile 32x64: two A row-fragments share four B fragments
//   -> 8 WMMA per K-step on 12 loads (1.5 loads/WMMA).
//   WG = 2x4 waves = 64x256 tile.  rope+silu epilogue, bf16 outputs.
// ---------------------------------------------------------------------------
__global__ __launch_bounds__(256) void qkv_kernel(
    const bf16_t* __restrict__ Xb,  const bf16_t* __restrict__ WTq,
    const bf16_t* __restrict__ WTk, const bf16_t* __restrict__ WTv,
    bf16_t* __restrict__ Qb, bf16_t* __restrict__ Kb, bf16_t* __restrict__ Vr)
{
  const int lane = threadIdx.x & 31;
  const int wave = threadIdx.x >> 5;
  const int wr   = wave >> 2;          // 0..1 row group
  const int wc   = wave & 3;           // 0..3 col group
  const int hi = lane >> 4;
  const int lo = lane & 15;

  const int row0 = blockIdx.x * 64 + wr * 32;     // global row tile (b*T+t)
  const int colg = blockIdx.y * 256 + wc * 64;    // 0..3071 across Q|K|V
  const int wsel = colg >> 10;                    // 0=Q 1=K 2=V
  const int n0   = colg & 1023;

  const bf16_t* WT = (wsel == 0) ? WTq : (wsel == 1) ? WTk : WTv;
  bf16_t*      Ob  = (wsel == 0) ? Qb  : (wsel == 1) ? Kb  : Vr;

  const bf16_t* Xrow0 = Xb + (size_t)(row0 + lo) * CIN + hi * 8;
  const bf16_t* Xrow1 = Xrow0 + 16 * CIN;
  const bf16_t* Wbase = WT + (size_t)(n0 + lo) * 1024 + hi * 8;

#define LD_A2(f, k)                                    \
  do {                                                 \
    (f)[0].q[0] = *(const uint4*)(Xrow0 + (k));        \
    (f)[0].q[1] = *(const uint4*)(Xrow0 + (k) + 16);   \
    (f)[1].q[0] = *(const uint4*)(Xrow1 + (k));        \
    (f)[1].q[1] = *(const uint4*)(Xrow1 + (k) + 16);   \
  } while (0)
#define LD_B4(f, k)                                    \
  _Pragma("unroll") for (int j2 = 0; j2 < 4; ++j2) {   \
    const bf16_t* Wp = Wbase + j2 * 16384 + (k);       \
    (f)[j2].q[0] = *(const uint4*)(Wp);                \
    (f)[j2].q[1] = *(const uint4*)(Wp + 16);           \
  }
#define WMMA8(a, b)                                                        \
  _Pragma("unroll") for (int ri = 0; ri < 2; ++ri)                         \
  _Pragma("unroll") for (int j2 = 0; j2 < 4; ++j2)                         \
    acc[ri][j2] = __builtin_amdgcn_wmma_f32_16x16x32_bf16(                 \
        false, (a)[ri].v, false, (b)[j2].v, (short)0, acc[ri][j2], false,  \
        false);

  v8f acc[2][4] = {};
  Frag a0[2], a1[2], b0[4], b1[4];
  LD_A2(a0, 0);
  LD_B4(b0, 0);
  for (int k0 = 0; k0 < CIN; k0 += 64) {
    LD_A2(a1, k0 + 32);
    LD_B4(b1, k0 + 32);
    WMMA8(a0, b0);
    if (k0 + 64 < CIN) {
      LD_A2(a0, k0 + 64);
      LD_B4(b0, k0 + 64);
    }
    WMMA8(a1, b1);
  }
#undef LD_A2
#undef LD_B4
#undef WMMA8

  // Epilogue: rope (Q/K, channels < 64) + silu, store bf16 row-major
  const bool rope_mat = (wsel < 2);
#pragma unroll
  for (int j2 = 0; j2 < 4; ++j2) {
    const int chb = n0 + 16 * j2;
    const bool dorope = rope_mat && (chb < 64);   // uniform per tile
    const int ch = chb + lo;
    const float e   = (float)(ch & 62) * (1.0f / 64.0f);
    const float inv = exp2f(-13.287712379549449f * e);  // 10000^(-(ch&~1)/64)
#pragma unroll
    for (int ri = 0; ri < 2; ++ri) {
#pragma unroll
      for (int r = 0; r < 8; ++r) {
        int m   = r + (hi << 3);
        int row = row0 + ri * 16 + m;
        int t   = row & (T_SEQ - 1);
        float x = acc[ri][j2][r];
        float partner = __shfl_xor(x, 1);   // channel ch^1, same row
        if (dorope) {
          float ang = (float)t * inv;
          float c = __cosf(ang), s = __sinf(ang);
          x = (ch & 1) ? (x * c + partner * s) : (x * c - partner * s);
        }
        x = silu_f(x);
        Ob[(size_t)row * 1024 + ch] = (bf16_t)x;
      }
    }
  }
}

// ---------------------------------------------------------------------------
// Kernel 2: fused decay-attention + group norm.
//   out[b,t,:] = GN( sum_{s>=t} gamma^(s-t) (Q[t].K[s]) V[s] )
//   One WG per 16-row t tile; Q tile staged in LDS once (reused by every
//   superstep / wave); stage-1 k-loop double-buffered (A via DScnt, B via
//   LOADcnt -> independent overlap).  V pre-transposed for contiguous B.
// ---------------------------------------------------------------------------
__global__ __launch_bounds__(256) void attn_kernel(
    const bf16_t* __restrict__ Qb, const bf16_t* __restrict__ Kb,
    const bf16_t* __restrict__ VT, const float* __restrict__ gnw,
    const float* __restrict__ gnb, float* __restrict__ out)
{
  __shared__ bf16_t Qs[16][1032];  // 2064B rows: 16B-aligned, 4-bank rotation
  __shared__ bf16_t S_s[16][136];  // 272B rows: 16B-aligned, 4-bank rotation

  const int lane = threadIdx.x & 31;
  const int wave = threadIdx.x >> 5;
  const int hi = lane >> 4;
  const int lo = lane & 15;

  const int tiles_per_seq = T_SEQ / 16;
  const int b  = blockIdx.x / tiles_per_seq;
  const int t0 = (blockIdx.x % tiles_per_seq) * 16;
  const size_t base = (size_t)b * T_SEQ * 1024;
  const size_t vtcol = (size_t)b * T_SEQ;        // VT is [1024][NB*T]

  const float L2G = -5.6465626e-3f;  // log2(255/256)

  // ---- stage Q[t0:+16, :] into LDS (once; reused every superstep)
  {
    int row = threadIdx.x >> 4;
    int ch0 = (threadIdx.x & 15) * 64;
    const bf16_t* src = Qb + base + (size_t)(t0 + row) * 1024 + ch0;
#pragma unroll
    for (int i = 0; i < 8; ++i)
      *(uint4*)(&Qs[row][ch0 + i * 8]) = *(const uint4*)(src + i * 8);
  }
  __syncthreads();

  v8f O[8] = {};  // 16 rows x 128 channels per wave

  for (int s0 = (t0 & ~127); s0 < T_SEQ; s0 += 128) {
    const int sw = s0 + wave * 16;  // this wave's 16-wide s sub-tile
    if (s0 + 128 < T_SEQ)           // prefetch next superstep's K rows
      __builtin_prefetch(Kb + base + (size_t)(s0 + 128 + wave * 16) * 1024, 0, 1);

    // ---- stage 1: S_w = Q[t0:+16, :] . K[sw:+16, :]^T   (K-dim 1024)
    const bf16_t* Krow = Kb + base + (size_t)(sw + lo) * 1024 + hi * 8;

#define LD_QA(f, k)                                              \
  do {                                                           \
    (f).q[0] = *(const uint4*)(&Qs[lo][(k) + hi * 8]);           \
    (f).q[1] = *(const uint4*)(&Qs[lo][(k) + 16 + hi * 8]);      \
  } while (0)
#define LD_KB(f, k)                                              \
  do {                                                           \
    (f).q[0] = *(const uint4*)(Krow + (k));                      \
    (f).q[1] = *(const uint4*)(Krow + (k) + 16);                 \
  } while (0)

    v8f Sacc = {};
    Frag a0, a1, c0, c1;
    LD_QA(a0, 0);
    LD_KB(c0, 0);
    for (int k0 = 0; k0 < 1024; k0 += 64) {
      LD_QA(a1, k0 + 32);
      LD_KB(c1, k0 + 32);
      Sacc = __builtin_amdgcn_wmma_f32_16x16x32_bf16(
          false, a0.v, false, c0.v, (short)0, Sacc, false, false);
      if (k0 + 64 < 1024) {
        LD_QA(a0, k0 + 64);
        LD_KB(c0, k0 + 64);
      }
      Sacc = __builtin_amdgcn_wmma_f32_16x16x32_bf16(
          false, a1.v, false, c1.v, (short)0, Sacc, false, false);
    }
#undef LD_QA
#undef LD_KB

    // decay mask gamma^(s-t) for s>=t, cvt to bf16, stage in LDS
#pragma unroll
    for (int r = 0; r < 8; ++r) {
      int m = r + (hi << 3);
      int t = t0 + m;
      int s = sw + lo;
      float d = (s >= t) ? exp2f((float)(s - t) * L2G) : 0.0f;
      S_s[m][wave * 16 + lo] = (bf16_t)(Sacc[r] * d);
    }
    __syncthreads();

    // ---- stage 2: O += S(16x128) . V[s0:+128, wave*128:+128]
#pragma unroll
    for (int kk = 0; kk < 4; ++kk) {
      Frag a;
      a.q[0] = *(const uint4*)(&S_s[lo][kk * 32 + hi * 8]);
      a.q[1] = *(const uint4*)(&S_s[lo][kk * 32 + 16 + hi * 8]);
#pragma unroll
      for (int j2 = 0; j2 < 8; ++j2) {
        // B(k,n) = VT[n][s]; per lane n fixed, two runs of 8 s (table order)
        const bf16_t* Vp = VT + (size_t)(wave * 128 + j2 * 16 + lo) *
                               (NB * T_SEQ) + vtcol + s0 + kk * 32 + hi * 8;
        Frag bv;
        bv.q[0] = *(const uint4*)(Vp);
        bv.q[1] = *(const uint4*)(Vp + 16);
        O[j2] = __builtin_amdgcn_wmma_f32_16x16x32_bf16(
            false, a.v, false, bv.v, (short)0, O[j2], false, false);
      }
    }
    __syncthreads();
  }

  // ---- fused group norm (32-ch groups; each wave owns 4 whole groups)
  const int chbase = wave * 128;
#pragma unroll
  for (int r = 0; r < 8; ++r) {
    int m = r + (hi << 3);
    size_t orow = base + (size_t)(t0 + m) * 1024;
#pragma unroll
    for (int p = 0; p < 4; ++p) {
      float x0 = O[2 * p][r];      // channel chbase+32p+lo
      float x1 = O[2 * p + 1][r];  // channel chbase+32p+16+lo
      float s1 = x0 + x1;
      float s2 = x0 * x0 + x1 * x1;
#pragma unroll
      for (int msk = 8; msk >= 1; msk >>= 1) {  // stays inside 16-lane half
        s1 += __shfl_xor(s1, msk);
        s2 += __shfl_xor(s2, msk);
      }
      float mu  = s1 * (1.0f / 32.0f);
      float var = s2 * (1.0f / 32.0f) - mu * mu;
      float inv = rsqrtf(var + 1e-6f);
      int c0 = chbase + 32 * p + lo;
      int c1 = c0 + 16;
      out[orow + c0] = (x0 - mu) * inv * gnw[c0] + gnb[c0];
      out[orow + c1] = (x1 - mu) * inv * gnw[c1] + gnb[c1];
    }
  }
}

// ---------------------------------------------------------------------------
extern "C" void kernel_launch(void* const* d_in, const int* in_sizes, int n_in,
                              void* d_out, int out_size, void* d_ws, size_t ws_size,
                              hipStream_t stream) {
  const float* X  = (const float*)d_in[0];
  const float* WQ = (const float*)d_in[1];
  const float* WK = (const float*)d_in[2];
  const float* WV = (const float*)d_in[3];
  const float* gw = (const float*)d_in[4];
  const float* gb = (const float*)d_in[5];
  float* out = (float*)d_out;

  const size_t QK = (size_t)NB * T_SEQ * 1024;   // 8M elems
  bf16_t* Xb  = (bf16_t*)d_ws;                   // 16 MB
  bf16_t* WTq = Xb  + QK;                        //  2 MB
  bf16_t* WTk = WTq + 1024 * 1024;               //  2 MB
  bf16_t* WTv = WTk + 1024 * 1024;               //  2 MB
  bf16_t* Qb  = WTv + 1024 * 1024;               // 16 MB
  bf16_t* Kb  = Qb  + QK;                        // 16 MB
  bf16_t* Vr  = Kb  + QK;                        // 16 MB
  bf16_t* VT  = Vr  + QK;                        // 16 MB (total 86 MB)

  // prep: bf16 conversions + weight transposes
  xcvt_kernel<<<dim3(QK / (256 * 4)), dim3(256), 0, stream>>>(X, Xb);
  trans_kernel<float><<<dim3(1024), dim3(256), 0, stream>>>(WQ, WTq, 1024, 1024);
  trans_kernel<float><<<dim3(1024), dim3(256), 0, stream>>>(WK, WTk, 1024, 1024);
  trans_kernel<float><<<dim3(1024), dim3(256), 0, stream>>>(WV, WTv, 1024, 1024);

  // QKV projection (rope+silu fused): 64x256 per WG
  dim3 g1(NB * T_SEQ / 64, 12);
  qkv_kernel<<<g1, dim3(256), 0, stream>>>(Xb, WTq, WTk, WTv, Qb, Kb, Vr);

  // V -> V^T so attention B-operands are contiguous
  trans_kernel<bf16_t><<<dim3((NB * T_SEQ / 32) * (1024 / 32)), dim3(256), 0,
                         stream>>>(Vr, VT, NB * T_SEQ, 1024);

  // decay attention + group norm
  attn_kernel<<<dim3(NB * (T_SEQ / 16)), dim3(256), 0, stream>>>(
      Qb, Kb, VT, gw, gb, out);
}
